// EncoderSAGPool_48275432407778
// MI455X (gfx1250) — matrix operand order repaired and checked by reading, not compiled, not run on hardware
//
#include <hip/hip_runtime.h>
#include <hip/hip_bf16.h>
#include <math.h>

// Problem sizes (fixed by the reference)
#define NN   8192      // B * NP nodes
#define BB   64        // batches
#define NPK  128       // nodes per batch
#define KSEL 26        // top-k
#define NP2  1664      // BB * KSEL  (= 13 * 128 = 26 * 64)
#define EMBD 512
#define WD0  300
#define WDP  320       // WD padded to multiple of 32

typedef __bf16 v16bf __attribute__((ext_vector_type(16)));
typedef float  v8f   __attribute__((ext_vector_type(8)));

static __device__ __forceinline__ unsigned short f2bf(float f) {
  unsigned int u = __float_as_uint(f);
  u += 0x7fffu + ((u >> 16) & 1u);          // round-to-nearest-even
  return (unsigned short)(u >> 16);
}
static __device__ __forceinline__ float bf2f(unsigned short h) {
  return __uint_as_float(((unsigned int)h) << 16);
}

// ---------------------------------------------------------------------------
// WMMA bf16 GEMM:  C(MxN) = A(MxK) @ B(KxN), fp32 accum.
// A row-major (MxK); B supplied TRANSPOSED row-major (BT: NxK) so both
// fragment reads are contiguous 16B vector LDS loads.
// Epilogue configuration is compile-time (branchless straight-line code):
//   RS: multiply by rowscale[m]; BIASF: add bias[n]; RELU; CF/CB: f32/bf16 out.
// Requires: M % 128 == 0, N % 64 == 0, K % 32 == 0 (guaranteed by caller).
// Block = 256 threads (8 wave32), tile 128x64, each wave 32x32 (2x2 WMMA),
// double-buffered LDS software pipeline, one barrier per K-step.
// ---------------------------------------------------------------------------
#define BM 128
#define BN 64
#define BK 32
#define LDP 40   // padded LDS row pitch (ushorts) to spread banks

template <bool RS, bool BIASF, bool RELU, bool CF, bool CB>
__global__ __launch_bounds__(256)
void k_gemm_bf16(const unsigned short* __restrict__ A, int lda,
                 const unsigned short* __restrict__ BT, int ldb,   // BT: N x K
                 float* __restrict__ Cf, int ldcf,
                 unsigned short* __restrict__ Cb, int ldcb,
                 const float* __restrict__ rowscale,
                 const float* __restrict__ bias,
                 int K)
{
  __shared__ __align__(16) unsigned short As[2][BM][LDP];
  __shared__ __align__(16) unsigned short Bs[2][BN][LDP];

  const int tid  = threadIdx.x;
  const int lane = tid & 31;
  const int wave = tid >> 5;
  const int wm   = (wave >> 1) * 32;     // 0,32,64,96
  const int wn   = (wave & 1) * 32;      // 0,32
  const int m0   = blockIdx.y * BM;
  const int n0   = blockIdx.x * BN;

  v8f acc[2][2];
  const v8f vzero = {0.f, 0.f, 0.f, 0.f, 0.f, 0.f, 0.f, 0.f};
#pragma unroll
  for (int i = 0; i < 2; ++i)
#pragma unroll
    for (int j = 0; j < 2; ++j) acc[i][j] = vzero;

  // staging maps: A tile 128(m) x 32(k); B tile 64(n) x 32(k)
  const int arow = tid >> 1;             // 0..127
  const int acol = (tid & 1) * 16;       // 0 or 16
  const int brow = tid >> 2;             // 0..63
  const int bcol = (tid & 3) * 8;        // 0,8,16,24

  const unsigned short* agp = A  + (size_t)(m0 + arow) * lda + acol;
  const unsigned short* bgp = BT + (size_t)(n0 + brow) * ldb + bcol;

  // prologue: stage tile kt=0 into buffer 0
  {
    uint4 ra0 = *(const uint4*)(agp);
    uint4 ra1 = *(const uint4*)(agp + 8);
    uint4 rb  = *(const uint4*)(bgp);
    *(uint4*)&As[0][arow][acol]     = ra0;
    *(uint4*)&As[0][arow][acol + 8] = ra1;
    *(uint4*)&Bs[0][brow][bcol]     = rb;
  }
  __syncthreads();

  const int am   = lane & 15;
  const int akhi = (lane >> 4) << 3;     // lanes 16..31 take K+8 groups
  const int bkb  = (lane >> 4) << 4;     // unused: B fragments use akhi (same k order as A)

  int p = 0;
  for (int kt = 0; kt < K; kt += BK) {
    const bool has_next = (kt + BK) < K;
    uint4 ra0, ra1, rb;
    if (has_next) {
      // issue next tile's global loads early; latency hides under WMMA
      ra0 = *(const uint4*)(agp + (kt + BK));
      ra1 = *(const uint4*)(agp + (kt + BK) + 8);
      rb  = *(const uint4*)(bgp + (kt + BK));
      if (kt + 2 * BK < K) {
        __builtin_prefetch(agp + (kt + 2 * BK), 0, 1);
        __builtin_prefetch(bgp + (kt + 2 * BK), 0, 1);
      }
    }

    // fragments (all contiguous 16B LDS vector loads)
    v16bf afrag[2], bfrag[2];
#pragma unroll
    for (int s = 0; s < 2; ++s) {
      union { v16bf v; uint4 q[2]; } au;
      const unsigned short* ap = &As[p][wm + s * 16 + am][0];
      au.q[0] = *(const uint4*)(ap + akhi);        // K = akhi .. akhi+7
      au.q[1] = *(const uint4*)(ap + akhi + 16);   // K = akhi+16 .. akhi+23
      afrag[s] = au.v;
      union { v16bf v; uint4 q[2]; } bu;
      const unsigned short* bp = &Bs[p][wn + s * 16 + am][0];
      bu.q[0] = *(const uint4*)(bp + akhi);        // K = akhi .. akhi+7
      bu.q[1] = *(const uint4*)(bp + akhi + 16);   // K = akhi+16 .. akhi+23
      bfrag[s] = bu.v;
    }

#pragma unroll
    for (int i = 0; i < 2; ++i)
#pragma unroll
      for (int j = 0; j < 2; ++j)
        acc[i][j] = __builtin_amdgcn_wmma_f32_16x16x32_bf16(
            false, afrag[i], false, bfrag[j], (short)0, acc[i][j], false, false);

    if (has_next) {
      *(uint4*)&As[p ^ 1][arow][acol]     = ra0;
      *(uint4*)&As[p ^ 1][arow][acol + 8] = ra1;
      *(uint4*)&Bs[p ^ 1][brow][bcol]     = rb;
    }
    __syncthreads();
    p ^= 1;
  }

  // Branchless epilogue (C layout: lane = N mod 16, VGPR r -> M = r + 8*(lane>=16))
  const int mhi = (lane >> 4) << 3;
  const int nlo = lane & 15;
  float bv[2];
#pragma unroll
  for (int j = 0; j < 2; ++j)
    bv[j] = BIASF ? bias[n0 + wn + j * 16 + nlo] : 0.0f;
#pragma unroll
  for (int i = 0; i < 2; ++i) {
#pragma unroll
    for (int r = 0; r < 8; ++r) {
      const int gm = m0 + wm + i * 16 + r + mhi;
      const float rsv = RS ? rowscale[gm] : 1.0f;
#pragma unroll
      for (int j = 0; j < 2; ++j) {
        const int gn = n0 + wn + j * 16 + nlo;
        float v = acc[i][j][r];
        if (RS)    v *= rsv;
        if (BIASF) v += bv[j];
        if (RELU)  v = fmaxf(v, 0.0f);
        if (CF) Cf[(size_t)gm * ldcf + gn] = v;
        if (CB) Cb[(size_t)gm * ldcb + gn] = f2bf(v);
      }
    }
  }
}

// ---------------------------------------------------------------------------
// Adjacency build:  Ah = (adj + adj^T > 0) + I, stored bf16 (values 0/1/2 exact)
// ---------------------------------------------------------------------------
__global__ __launch_bounds__(256)
void k_build_adj(const float* __restrict__ adj, unsigned short* __restrict__ Ahb)
{
  __shared__ float t0[64][65];
  __shared__ float t1[64][65];
  const int i0 = blockIdx.y * 64;
  const int j0 = blockIdx.x * 64;
  const int tid = threadIdx.x;
#pragma unroll
  for (int e = 0; e < 16; ++e) {
    const int lin = e * 256 + tid;
    const int r = lin >> 6, c = lin & 63;
    t0[r][c] = adj[(size_t)(i0 + r) * NN + (j0 + c)];
    t1[r][c] = adj[(size_t)(j0 + r) * NN + (i0 + c)];
  }
  __syncthreads();
#pragma unroll
  for (int e = 0; e < 16; ++e) {
    const int lin = e * 256 + tid;
    const int r = lin >> 6, c = lin & 63;
    float ah = ((t0[r][c] + t1[c][r]) > 0.0f) ? 1.0f : 0.0f;
    if (i0 + r == j0 + c) ah += 1.0f;   // +I
    Ahb[(size_t)(i0 + r) * NN + (j0 + c)] = f2bf(ah);
  }
}

// dinv[i] = rsqrt(sum_j Ah[i][j])   (one block per row)
__global__ __launch_bounds__(256)
void k_rowsum_dinv(const unsigned short* __restrict__ Ah, int ncols,
                   float* __restrict__ dinv)
{
  __shared__ float red[256];
  const int i = blockIdx.x, t = threadIdx.x;
  float s = 0.f;
  for (int j = t; j < ncols; j += 256) s += bf2f(Ah[(size_t)i * ncols + j]);
  red[t] = s; __syncthreads();
  for (int st = 128; st > 0; st >>= 1) {
    if (t < st) red[t] += red[t + st];
    __syncthreads();
  }
  if (t == 0) dinv[i] = rsqrtf(red[0]);
}

// Embedding gather -> bf16, K padded 300->320 with zeros (one block per node)
__global__ __launch_bounds__(256)
void k_embed(const int* __restrict__ nodes, const float* __restrict__ tab,
             unsigned short* __restrict__ Xb)
{
  const int i = blockIdx.x;
  const int node = nodes[i];
  for (int c = threadIdx.x; c < WDP; c += 256)
    Xb[(size_t)i * WDP + c] = (c < WD0) ? f2bf(tab[(size_t)node * WD0 + c])
                                        : (unsigned short)0;
}

// Weight convert fp32 (Kr x Nc) -> bf16 TRANSPOSED (Nc x Kp), zero K padding
__global__ __launch_bounds__(256)
void k_cvt_padT(const float* __restrict__ src, unsigned short* __restrict__ dst,
                int Kr, int Kp, int Nc)
{
  const long idx = (long)blockIdx.x * 256 + threadIdx.x;
  if (idx >= (long)Nc * Kp) return;
  const int n = (int)(idx / Kp);
  const int k = (int)(idx % Kp);
  dst[idx] = (k < Kr) ? f2bf(src[(size_t)k * Nc + n]) : (unsigned short)0;
}

// Zbt[c][i] = bf16(dinv[i] * Y[i][c])  — tiled transpose, both sides coalesced.
// Y: M x EMBD fp32 ; Zbt: EMBD x M bf16. grid (M/64, EMBD/64)
__global__ __launch_bounds__(256)
void k_scale_cvtT(const float* __restrict__ Y, const float* __restrict__ dinv,
                  unsigned short* __restrict__ Zbt, int M)
{
  __shared__ float t[64][65];
  const int i0 = blockIdx.x * 64;
  const int c0 = blockIdx.y * 64;
  const int tid = threadIdx.x;
#pragma unroll
  for (int e = 0; e < 16; ++e) {
    const int lin = e * 256 + tid;
    const int r = lin >> 6, c = lin & 63;
    t[r][c] = Y[(size_t)(i0 + r) * EMBD + (c0 + c)];
  }
  __syncthreads();
#pragma unroll
  for (int e = 0; e < 16; ++e) {
    const int lin = e * 256 + tid;
    const int r = lin >> 6, c = lin & 63;   // r: local col of Y, c: local row of Y
    Zbt[(size_t)(c0 + r) * M + (i0 + c)] = f2bf(dinv[i0 + c] * t[c][r]);
  }
}

// z[i] = dinv[i] * dot(x[i,:], Ws)   (one block per row)
__global__ __launch_bounds__(256)
void k_xw_scale(const float* __restrict__ x, const float* __restrict__ Ws,
                const float* __restrict__ dinv, float* __restrict__ z)
{
  __shared__ float red[256];
  const int i = blockIdx.x, t = threadIdx.x;
  float p = 0.f;
  for (int c = t; c < EMBD; c += 256) p += x[(size_t)i * EMBD + c] * Ws[c];
  red[t] = p; __syncthreads();
  for (int st = 128; st > 0; st >>= 1) {
    if (t < st) red[t] += red[t + st];
    __syncthreads();
  }
  if (t == 0) z[i] = dinv[i] * red[0];
}

// score[i] = dinv[i] * dot(Ah[i,:], z) + bs   (one block per row)
__global__ __launch_bounds__(256)
void k_score(const unsigned short* __restrict__ Ahb, const float* __restrict__ z,
             const float* __restrict__ dinv, const float* __restrict__ bs,
             float* __restrict__ score)
{
  __shared__ float red[256];
  const int i = blockIdx.x, t = threadIdx.x;
  float p = 0.f;
  for (int j = t; j < NN; j += 256) p += bf2f(Ahb[(size_t)i * NN + j]) * z[j];
  red[t] = p; __syncthreads();
  for (int st = 128; st > 0; st >>= 1) {
    if (t < st) red[t] += red[t + st];
    __syncthreads();
  }
  if (t == 0) score[i] = dinv[i] * red[0] + bs[0];
}

// Per-batch top-26 of 128 (one block per batch, parallel arg-max selection)
__global__ __launch_bounds__(128)
void k_topk(const float* __restrict__ score, int* __restrict__ perm)
{
  __shared__ float vm[NPK];
  __shared__ float v[NPK];
  __shared__ int   ix[NPK];
  const int b = blockIdx.x, t = threadIdx.x;
  vm[t] = score[b * NPK + t];
  __syncthreads();
  for (int k = 0; k < KSEL; ++k) {
    v[t] = vm[t]; ix[t] = t;
    __syncthreads();
    for (int s = 64; s > 0; s >>= 1) {
      if (t < s) {
        if (v[t + s] > v[t] || (v[t + s] == v[t] && ix[t + s] < ix[t])) {
          v[t] = v[t + s]; ix[t] = ix[t + s];
        }
      }
      __syncthreads();
    }
    if (t == 0) { perm[b * KSEL + k] = b * NPK + ix[0]; vm[ix[0]] = -INFINITY; }
    __syncthreads();
  }
}

// xp[r][c] = bf16( x[perm[r]][c] * tanh(score[perm[r]]) )
__global__ __launch_bounds__(256)
void k_gather_xp(const float* __restrict__ x, const float* __restrict__ score,
                 const int* __restrict__ perm, unsigned short* __restrict__ xpb)
{
  const int r = blockIdx.x;
  const int p = perm[r];
  const float s = tanhf(score[p]);
  for (int c = threadIdx.x; c < EMBD; c += 256)
    xpb[(size_t)r * EMBD + c] = f2bf(x[(size_t)p * EMBD + c] * s);
}

// Ahp = Ah[perm, perm]  (exactly equals Ap + I), plus dinvp = rsqrt(rowsum)
__global__ __launch_bounds__(256)
void k_build_ahp(const unsigned short* __restrict__ Ahb, const int* __restrict__ perm,
                 unsigned short* __restrict__ Ahpb, float* __restrict__ dinvp)
{
  __shared__ int   pv[NP2];
  __shared__ float red[256];
  const int u = blockIdx.x, t = threadIdx.x;
  const int pu = perm[u];
  for (int v = t; v < NP2; v += 256) pv[v] = perm[v];
  __syncthreads();
  float s = 0.f;
  for (int v = t; v < NP2; v += 256) {
    const unsigned short a = Ahb[(size_t)pu * NN + pv[v]];
    Ahpb[(size_t)u * NP2 + v] = a;
    s += bf2f(a);
  }
  red[t] = s; __syncthreads();
  for (int st = 128; st > 0; st >>= 1) {
    if (t < st) red[t] += red[t + st];
    __syncthreads();
  }
  if (t == 0) dinvp[u] = rsqrtf(red[0]);
}

// Per-batch mean over cnt rows -> gdst[b][c], grid (BB, EMBD/256)
__global__ __launch_bounds__(256)
void k_mean(const float* __restrict__ x, int cnt, float* __restrict__ gdst, int gld)
{
  const int b = blockIdx.x;
  const int c = blockIdx.y * 256 + threadIdx.x;
  float s = 0.f;
  for (int r = 0; r < cnt; ++r) s += x[(size_t)(b * cnt + r) * EMBD + c];
  gdst[(size_t)b * gld + c] = s / (float)cnt;
}

// feat = GF(64x1024) @ Wo(1024x512) + bo, grid (64, 2)
__global__ __launch_bounds__(256)
void k_fc(const float* __restrict__ GF, const float* __restrict__ Wo,
          const float* __restrict__ bo, float* __restrict__ feat)
{
  const int b = blockIdx.x;
  const int c = blockIdx.y * 256 + threadIdx.x;
  float s = bo[c];
  for (int k = 0; k < 2 * EMBD; ++k)
    s += GF[(size_t)b * (2 * EMBD) + k] * Wo[(size_t)k * EMBD + c];
  feat[(size_t)b * EMBD + c] = s;
}

// out = feat / (||feat||_2 + 1e-10), one block per row
__global__ __launch_bounds__(256)
void k_norm(const float* __restrict__ feat, float* __restrict__ out)
{
  __shared__ float red[256];
  const int b = blockIdx.x, t = threadIdx.x;
  float s = 0.f;
  for (int c = t; c < EMBD; c += 256) {
    const float f = feat[(size_t)b * EMBD + c];
    s += f * f;
  }
  red[t] = s; __syncthreads();
  for (int st = 128; st > 0; st >>= 1) {
    if (t < st) red[t] += red[t + st];
    __syncthreads();
  }
  const float inv = 1.0f / (sqrtf(red[0]) + 1e-10f);
  for (int c = t; c < EMBD; c += 256)
    out[(size_t)b * EMBD + c] = feat[(size_t)b * EMBD + c] * inv;
}

// ---------------------------------------------------------------------------
extern "C" void kernel_launch(void* const* d_in, const int* in_sizes, int n_in,
                              void* d_out, int out_size, void* d_ws, size_t ws_size,
                              hipStream_t stream)
{
  const int*   nodes = (const int*)  d_in[0];
  const float* adj   = (const float*)d_in[1];
  // d_in[2] = batch (implied by layout, unused)
  const float* tab   = (const float*)d_in[3];
  const float* W1  = (const float*)d_in[4];  const float* b1  = (const float*)d_in[5];
  const float* W2  = (const float*)d_in[6];  const float* b2  = (const float*)d_in[7];
  const float* Wl  = (const float*)d_in[8];  const float* bl  = (const float*)d_in[9];
  const float* Ws  = (const float*)d_in[10]; const float* bs  = (const float*)d_in[11];
  const float* Wf1 = (const float*)d_in[12]; const float* bf1 = (const float*)d_in[13];
  const float* Wf2 = (const float*)d_in[14]; const float* bf2 = (const float*)d_in[15];
  const float* Wfl = (const float*)d_in[16]; const float* bfl = (const float*)d_in[17];
  const float* Wo  = (const float*)d_in[18]; const float* bo  = (const float*)d_in[19];
  float* out = (float*)d_out;

  // Bump allocator over d_ws (256B aligned slabs)
  char* wp = (char*)d_ws;
  auto alloc = [&](size_t bytes) -> void* {
    void* r = (void*)wp;
    wp += (bytes + 255) & ~(size_t)255;
    return r;
  };
  unsigned short* Ahb   = (unsigned short*)alloc((size_t)NN * NN * 2);   // 134 MB -> L2-resident
  float*          dinv  = (float*)         alloc((size_t)NN * 4);
  unsigned short* Xb    = (unsigned short*)alloc((size_t)NN * WDP * 2);
  unsigned short* W1b   = (unsigned short*)alloc((size_t)WDP * EMBD * 2);      // 512 x 320 (T)
  unsigned short* W2b   = (unsigned short*)alloc((size_t)EMBD * EMBD * 2);     // 512 x 512 (T)
  unsigned short* Wlb   = (unsigned short*)alloc((size_t)2 * EMBD * EMBD * 2); // 512 x 1024 (T)
  unsigned short* Wf1b  = (unsigned short*)alloc((size_t)EMBD * EMBD * 2);
  unsigned short* Wf2b  = (unsigned short*)alloc((size_t)EMBD * EMBD * 2);
  unsigned short* Wflb  = (unsigned short*)alloc((size_t)2 * EMBD * EMBD * 2);
  float*          Ybuf  = (float*)         alloc((size_t)NN * EMBD * 4);  // reused
  unsigned short* Zbt   = (unsigned short*)alloc((size_t)NN * EMBD * 2);  // EMBD x M (T), reused
  unsigned short* XC    = (unsigned short*)alloc((size_t)NN * 2 * EMBD * 2); // [x1|x2] bf16
  float*          xs    = (float*)         alloc((size_t)NN * EMBD * 4);  // stage-1 x fp32
  float*          zvec  = (float*)         alloc((size_t)NN * 4);
  float*          score = (float*)         alloc((size_t)NN * 4);
  int*            perm  = (int*)           alloc((size_t)NP2 * 4);
  unsigned short* xpb   = (unsigned short*)alloc((size_t)NP2 * EMBD * 2);
  unsigned short* Ahpb  = (unsigned short*)alloc((size_t)NP2 * NP2 * 2);
  float*          dinvp = (float*)         alloc((size_t)NP2 * 4);
  unsigned short* XCF   = (unsigned short*)alloc((size_t)NP2 * 2 * EMBD * 2);
  float*          xf    = (float*)         alloc((size_t)NP2 * EMBD * 4);
  float*          GF    = (float*)         alloc((size_t)BB * 2 * EMBD * 4);
  float*          feat  = (float*)         alloc((size_t)BB * EMBD * 4);

  // GEMM dispatch: four compile-time epilogue variants
  // plain: fp32 out only
  auto gemm_plain = [&](const unsigned short* A, int lda, const unsigned short* Bt, int ldb,
                        float* Cf, int M, int Kc) {
    dim3 g(EMBD / BN, M / BM);
    k_gemm_bf16<false, false, false, true, false><<<g, 256, 0, stream>>>(
        A, lda, Bt, ldb, Cf, EMBD, nullptr, 0, nullptr, nullptr, Kc);
  };
  // propagate: rowscale + bias + relu -> bf16 out (into concat buffer)
  auto gemm_prop = [&](const unsigned short* A, int lda, const unsigned short* Bt, int ldb,
                       unsigned short* Cb, const float* rs, const float* bias,
                       int M, int Kc) {
    dim3 g(EMBD / BN, M / BM);
    k_gemm_bf16<true, true, true, false, true><<<g, 256, 0, stream>>>(
        A, lda, Bt, ldb, nullptr, 0, Cb, 2 * EMBD, rs, bias, Kc);
  };
  // concat head: bias (+ optional relu) -> fp32 out
  auto gemm_bias_relu = [&](const unsigned short* A, int lda, const unsigned short* Bt, int ldb,
                            float* Cf, const float* bias, int M, int Kc) {
    dim3 g(EMBD / BN, M / BM);
    k_gemm_bf16<false, true, true, true, false><<<g, 256, 0, stream>>>(
        A, lda, Bt, ldb, Cf, EMBD, nullptr, 0, nullptr, bias, Kc);
  };
  auto gemm_bias = [&](const unsigned short* A, int lda, const unsigned short* Bt, int ldb,
                       float* Cf, const float* bias, int M, int Kc) {
    dim3 g(EMBD / BN, M / BM);
    k_gemm_bf16<false, true, false, true, false><<<g, 256, 0, stream>>>(
        A, lda, Bt, ldb, Cf, EMBD, nullptr, 0, nullptr, bias, Kc);
  };

  auto cvtT = [&](const float* src, unsigned short* dst, int Kr, int Kp, int Nc) {
    long tot = (long)Nc * Kp;
    k_cvt_padT<<<(unsigned)((tot + 255) / 256), 256, 0, stream>>>(src, dst, Kr, Kp, Nc);
  };
  auto sclT = [&](const float* Y, const float* dv, unsigned short* Z, int M) {
    k_scale_cvtT<<<dim3(M / 64, EMBD / 64), 256, 0, stream>>>(Y, dv, Z, M);
  };

  // --- adjacency: Ah = (adj + adj^T > 0) + I (bf16 exact), dinv = rsqrt(rowsum)
  k_build_adj<<<dim3(NN / 64, NN / 64), 256, 0, stream>>>(adj, Ahb);
  k_rowsum_dinv<<<NN, 256, 0, stream>>>(Ahb, NN, dinv);

  // --- inputs -> bf16 (weights transposed for WMMA B-fragment locality)
  k_embed<<<NN, 256, 0, stream>>>(nodes, tab, Xb);
  cvtT(W1,  W1b,  WD0,      WDP,      EMBD);
  cvtT(W2,  W2b,  EMBD,     EMBD,     EMBD);
  cvtT(Wl,  Wlb,  2 * EMBD, 2 * EMBD, EMBD);
  cvtT(Wf1, Wf1b, EMBD,     EMBD,     EMBD);
  cvtT(Wf2, Wf2b, EMBD,     EMBD,     EMBD);
  cvtT(Wfl, Wflb, 2 * EMBD, 2 * EMBD, EMBD);

  // --- stage-1 GNN block:  x1 = relu(D Ah D (x W1) + b1), x2 = relu(... W2 ...),
  //     x = relu([x1|x2] Wl + bl)
  gemm_plain(Xb, WDP, W1b, WDP, Ybuf, NN, WDP);
  sclT(Ybuf, dinv, Zbt, NN);
  gemm_prop(Ahb, NN, Zbt, NN, XC, dinv, b1, NN, NN);            // x1 -> XC[:, :512]
  gemm_plain(XC, 2 * EMBD, W2b, EMBD, Ybuf, NN, EMBD);
  sclT(Ybuf, dinv, Zbt, NN);
  gemm_prop(Ahb, NN, Zbt, NN, XC + EMBD, dinv, b2, NN, NN);     // x2 -> XC[:, 512:]
  gemm_bias_relu(XC, 2 * EMBD, Wlb, 2 * EMBD, xs, bl, NN, 2 * EMBD);

  // --- g1 = per-batch mean
  k_mean<<<dim3(BB, EMBD / 256), 256, 0, stream>>>(xs, NPK, GF, 2 * EMBD);

  // --- score = D Ah D (x Ws) + bs ; top-k ; pooled gather
  k_xw_scale<<<NN, 256, 0, stream>>>(xs, Ws, dinv, zvec);
  k_score<<<NN, 256, 0, stream>>>(Ahb, zvec, dinv, bs, score);
  k_topk<<<BB, 128, 0, stream>>>(score, perm);
  k_gather_xp<<<NP2, 256, 0, stream>>>(xs, score, perm, xpb);
  k_build_ahp<<<NP2, 256, 0, stream>>>(Ahb, perm, Ahpb, dinvp);

  // --- stage-2 GNN block on 1664 pooled nodes (no trailing relu)
  gemm_plain(xpb, EMBD, Wf1b, EMBD, Ybuf, NP2, EMBD);
  sclT(Ybuf, dinvp, Zbt, NP2);
  gemm_prop(Ahpb, NP2, Zbt, NP2, XCF, dinvp, bf1, NP2, NP2);
  gemm_plain(XCF, 2 * EMBD, Wf2b, EMBD, Ybuf, NP2, EMBD);
  sclT(Ybuf, dinvp, Zbt, NP2);
  gemm_prop(Ahpb, NP2, Zbt, NP2, XCF + EMBD, dinvp, bf2, NP2, NP2);
  gemm_bias(XCF, 2 * EMBD, Wflb, 2 * EMBD, xf, bfl, NP2, 2 * EMBD);

  // --- g2 = per-batch mean over K=26
  k_mean<<<dim3(BB, EMBD / 256), 256, 0, stream>>>(xf, KSEL, GF + EMBD, 2 * EMBD);

  // --- head: feat = [g1|g2] Wo + bo ; L2 normalize
  k_fc<<<dim3(BB, EMBD / 256), 256, 0, stream>>>(GF, Wo, bo, feat);
  k_norm<<<BB, 256, 0, stream>>>(feat, out);
}
